// MinGRULM_51539608289
// MI455X (gfx1250) — matrix operand order repaired and checked by reading, not compiled, not run on hardware
//
#include <hip/hip_runtime.h>
#include <hip/hip_bf16.h>

// ---------------- problem constants ----------------
#define VOCAB   256
#define DMODEL  512
#define NLAYER  12
#define BATCH   8
#define SEQ     8192
#define NROWS   (BATCH * SEQ)          // 65536 GEMM rows
#define NCHUNK  32
#define CHUNK   (SEQ / NCHUNK)         // 256

typedef __attribute__((ext_vector_type(16))) _Float16 v16h;
typedef __attribute__((ext_vector_type(8)))  float    v8f;
typedef int v4i __attribute__((vector_size(16)));     // matches builtin param type

// ---------------- CDNA5 async global->LDS helpers ----------------
__device__ __forceinline__ void async_ld_b128(const void* gptr, void* sptr) {
#if __has_builtin(__builtin_amdgcn_global_load_async_to_lds_b128)
    __builtin_amdgcn_global_load_async_to_lds_b128(
        (__attribute__((address_space(1))) v4i*)gptr,
        (__attribute__((address_space(3))) v4i*)sptr,
        0 /*offset*/, 0 /*cpol*/);
#else
    unsigned lds = (unsigned)(size_t)(__attribute__((address_space(3))) void*)sptr;
    asm volatile("global_load_async_to_lds_b128 %0, %1, off"
                 :: "v"(lds), "v"(gptr) : "memory");
#endif
}

#if __has_builtin(__builtin_amdgcn_s_wait_asynccnt)
#define WAIT_ASYNC(n) __builtin_amdgcn_s_wait_asynccnt(n)
#else
#define WAIT_ASYNC(n) asm volatile("s_wait_asynccnt %0" :: "i"(n) : "memory")
#endif

// ---------------- small prep kernels ----------------

// W (L, D, 2D) f32 -> WT (L, 2D, D) f16  (n-major, k contiguous == B^T layout)
__global__ void mingru_convert_w(const float* __restrict__ W, _Float16* __restrict__ WT) {
    int tid = blockIdx.x * blockDim.x + threadIdx.x;
    if (tid >= NLAYER * 2 * DMODEL * DMODEL) return;
    int k = tid % DMODEL;
    int n = (tid / DMODEL) % (2 * DMODEL);
    int l = tid / (DMODEL * 2 * DMODEL);
    WT[tid] = (_Float16)W[((size_t)l * DMODEL + k) * (2 * DMODEL) + n];
}

// emb (V, D) f32 -> f16 (already B^T layout for logits GEMM)
__global__ void mingru_convert_emb(const float* __restrict__ E, _Float16* __restrict__ E16) {
    int tid = blockIdx.x * blockDim.x + threadIdx.x;
    if (tid >= VOCAB * DMODEL) return;
    E16[tid] = (_Float16)E[tid];
}

// x[row, :] = emb[tokens[row], :]   (float4 copies)
__global__ void mingru_embed(const int* __restrict__ tok, const float* __restrict__ E,
                             float* __restrict__ x) {
    int tid = blockIdx.x * blockDim.x + threadIdx.x;   // NROWS * D/4
    int row = tid >> 7;                // D/4 = 128
    int d4  = tid & 127;
    int t   = tok[row];
    ((float4*)x)[(size_t)row * 128 + d4] =
        ((const float4*)E)[(size_t)t * 128 + d4];
}

// ---------------- layernorm -> f16 ----------------
__global__ __launch_bounds__(256)
void mingru_ln_f16(const float* __restrict__ x, const float* __restrict__ g,
                   const float* __restrict__ b, _Float16* __restrict__ out) {
    int row = blockIdx.x;
    int t   = threadIdx.x;
    const float* xr = x + (size_t)row * DMODEL;
    float v0 = xr[t];
    float v1 = xr[t + 256];
    __shared__ float s_sum[256];
    __shared__ float s_sq[256];
    s_sum[t] = v0 + v1;
    s_sq[t]  = v0 * v0 + v1 * v1;
    __syncthreads();
    for (int o = 128; o > 0; o >>= 1) {
        if (t < o) { s_sum[t] += s_sum[t + o]; s_sq[t] += s_sq[t + o]; }
        __syncthreads();
    }
    float mu  = s_sum[0] * (1.0f / DMODEL);
    float var = s_sq[0] * (1.0f / DMODEL) - mu * mu;
    float rs  = rsqrtf(var + 1e-5f);
    _Float16* orow = out + (size_t)row * DMODEL;
    orow[t]       = (_Float16)((v0 - mu) * rs * g[t]       + b[t]);
    orow[t + 256] = (_Float16)((v1 - mu) * rs * g[t + 256] + b[t + 256]);
}

// ---------------- WMMA GEMM: C(MxN) = A(MxK,f16) * BT(NxK,f16)^T ----------------
// Double-buffered LDS, async global->LDS tile copies (ASYNCcnt), wave32 WMMA.
#define BM 128
#define BN 128
#define BK 32
#define LDSP 40   // padded LDS row stride in halves (conflict-free b128 reads)

union FragU { v16h v; uint4 q[2]; };

__device__ __forceinline__ v16h ld_frag(const _Float16* p0, const _Float16* p1) {
    FragU u;
    u.q[0] = *(const uint4*)p0;
    u.q[1] = *(const uint4*)p1;
    return u.v;
}

__global__ __launch_bounds__(256)
void mingru_gemm_f16(const _Float16* __restrict__ A, const _Float16* __restrict__ BT,
                     float* __restrict__ C, int M, int N, int K) {
    __shared__ _Float16 sA[2][BM][LDSP];
    __shared__ _Float16 sB[2][BN][LDSP];

    int tid  = threadIdx.x;
    int lane = tid & 31;
    int wave = tid >> 5;
    int wm   = wave & 3;     // 4 wave-rows of 32
    int wn   = wave >> 2;    // 2 wave-cols of 64
    int bm   = blockIdx.y * BM;
    int bn   = blockIdx.x * BN;

    int ldr     = tid >> 1;            // row 0..127 this thread copies
    int lds_sel = (tid & 1) * 16;      // half-offset 0 or 16

    const _Float16* gA = A  + (size_t)(bm + ldr) * K + lds_sel;
    const _Float16* gB = BT + (size_t)(bn + ldr) * K + lds_sel;

    // prologue: async-copy tile 0 into buffer 0
    async_ld_b128(gA,     &sA[0][ldr][lds_sel]);
    async_ld_b128(gA + 8, &sA[0][ldr][lds_sel + 8]);
    async_ld_b128(gB,     &sB[0][ldr][lds_sel]);
    async_ld_b128(gB + 8, &sB[0][ldr][lds_sel + 8]);

    v8f acc[2][4];
    v8f zero = {0.f, 0.f, 0.f, 0.f, 0.f, 0.f, 0.f, 0.f};
#pragma unroll
    for (int f = 0; f < 2; ++f)
#pragma unroll
        for (int g2 = 0; g2 < 4; ++g2) acc[f][g2] = zero;

    int arow = wm * 32 + (lane & 15);
    int akb  = (lane >> 4) * 8;    // A: K chunk base (ISA f16 A layout)
    int bkb  = (lane >> 4) * 8;    // B: K chunk base (same table order as A)

    int ntiles = K / BK;
    for (int t = 0; t < ntiles; ++t) {
        int cur = t & 1;
        int nxt = cur ^ 1;
        if (t + 1 < ntiles) {
            const _Float16* nA = gA + (size_t)(t + 1) * BK;
            const _Float16* nB = gB + (size_t)(t + 1) * BK;
            async_ld_b128(nA,     &sA[nxt][ldr][lds_sel]);
            async_ld_b128(nA + 8, &sA[nxt][ldr][lds_sel + 8]);
            async_ld_b128(nB,     &sB[nxt][ldr][lds_sel]);
            async_ld_b128(nB + 8, &sB[nxt][ldr][lds_sel + 8]);
            WAIT_ASYNC(4);   // tile t's 4 copies complete (in-order)
        } else {
            WAIT_ASYNC(0);
        }
        __syncthreads();

        v16h af[2], bf[4];
        af[0] = ld_frag(&sA[cur][arow][akb],      &sA[cur][arow][akb + 16]);
        af[1] = ld_frag(&sA[cur][arow + 16][akb], &sA[cur][arow + 16][akb + 16]);
#pragma unroll
        for (int g2 = 0; g2 < 4; ++g2) {
            int nrow = wn * 64 + g2 * 16 + (lane & 15);
            bf[g2] = ld_frag(&sB[cur][nrow][bkb], &sB[cur][nrow][bkb + 16]);
        }
#pragma unroll
        for (int f = 0; f < 2; ++f)
#pragma unroll
            for (int g2 = 0; g2 < 4; ++g2)
                acc[f][g2] = __builtin_amdgcn_wmma_f32_16x16x32_f16(
                    false, af[f], false, bf[g2], (short)0, acc[f][g2], false, false);
        __syncthreads();
    }

    // C layout: lane<16 -> n=lane, rows m = r; lane>=16 -> n=lane-16, rows m = r+8
    int crow = bm + wm * 32 + (lane >> 4) * 8;
    int ccol = bn + wn * 64 + (lane & 15);
#pragma unroll
    for (int f = 0; f < 2; ++f)
#pragma unroll
        for (int g2 = 0; g2 < 4; ++g2)
#pragma unroll
            for (int r = 0; r < 8; ++r)
                C[(size_t)(crow + f * 16 + r) * N + (ccol + g2 * 16)] = acc[f][g2][r];
}

// ---------------- gating math ----------------
__device__ __forceinline__ void gate_ab(float hid, float gate, float& a, float& bt) {
    a = 1.0f / (1.0f + __expf(gate));                        // exp(-softplus(gate))
    float z  = 1.0f / (1.0f + __expf(-gate));                // sigmoid(gate)
    float gh = (hid >= 0.0f) ? (hid + 0.5f)
                             : 1.0f / (1.0f + __expf(-hid)); // g(hidden)
    bt = z * gh;
}

// ---------------- chunked scan: h_t = a_t*h_{t-1} + b_t ----------------
// phase 1: per (b, chunk, d) compute (prod a, local h end)
__global__ __launch_bounds__(256)
void mingru_scan_p1(const float* __restrict__ hg, float* __restrict__ cA,
                    float* __restrict__ cH) {
    int tid   = blockIdx.x * blockDim.x + threadIdx.x;   // BATCH*NCHUNK*DMODEL
    int d     = tid & (DMODEL - 1);
    int chunk = (tid >> 9) & (NCHUNK - 1);
    int b     = tid >> 14;
    size_t base = ((size_t)b * SEQ + (size_t)chunk * CHUNK) * (2 * DMODEL) + d;
    float Ap = 1.0f, H = 0.0f;
    for (int s = 0; s < CHUNK; ++s) {
        __builtin_prefetch(&hg[base + (size_t)8 * 2 * DMODEL], 0, 1);
        float hid  = hg[base];
        float gate = hg[base + DMODEL];
        float a, bt; gate_ab(hid, gate, a, bt);
        H  = a * H + bt;
        Ap = Ap * a;
        base += 2 * DMODEL;
    }
    cA[tid] = Ap;
    cH[tid] = H;
}

// phase 2: scan carries across chunks per (b, d); store each chunk's init h
__global__ __launch_bounds__(256)
void mingru_scan_p2(const float* __restrict__ cA, const float* __restrict__ cH,
                    float* __restrict__ hinit) {
    int tid = blockIdx.x * blockDim.x + threadIdx.x;     // BATCH*DMODEL
    int d   = tid & (DMODEL - 1);
    int b   = tid >> 9;
    float h = 0.0f;
    for (int k = 0; k < NCHUNK; ++k) {
        int ci = (b * NCHUNK + k) * DMODEL + d;
        hinit[ci] = h;
        h = cA[ci] * h + cH[ci];
    }
}

// phase 3: replay chunk with correct init, fuse residual x += h
__global__ __launch_bounds__(256)
void mingru_scan_p3(const float* __restrict__ hg, const float* __restrict__ hinit,
                    float* __restrict__ x) {
    int tid   = blockIdx.x * blockDim.x + threadIdx.x;
    int d     = tid & (DMODEL - 1);
    int chunk = (tid >> 9) & (NCHUNK - 1);
    int b     = tid >> 14;
    size_t base = ((size_t)b * SEQ + (size_t)chunk * CHUNK) * (2 * DMODEL) + d;
    size_t xb   = ((size_t)b * SEQ + (size_t)chunk * CHUNK) * DMODEL + d;
    float h = hinit[tid];
    for (int s = 0; s < CHUNK; ++s) {
        __builtin_prefetch(&hg[base + (size_t)8 * 2 * DMODEL], 0, 1);
        float hid  = hg[base];
        float gate = hg[base + DMODEL];
        float a, bt; gate_ab(hid, gate, a, bt);
        h = a * h + bt;
        x[xb] += h;
        base += 2 * DMODEL;
        xb   += DMODEL;
    }
}

// ---------------- launcher ----------------
extern "C" void kernel_launch(void* const* d_in, const int* in_sizes, int n_in,
                              void* d_out, int out_size, void* d_ws, size_t ws_size,
                              hipStream_t stream) {
    const int*   tokens = (const int*)  d_in[0];
    const float* emb    = (const float*)d_in[1];
    const float* ln_g   = (const float*)d_in[2];
    const float* ln_b   = (const float*)d_in[3];
    const float* W      = (const float*)d_in[4];
    const float* norm_g = (const float*)d_in[5];
    const float* norm_b = (const float*)d_in[6];
    float* logits = (float*)d_out;

    // workspace layout (bytes, all 256-aligned)
    char*  ws   = (char*)d_ws;
    size_t off  = 0;
    float*    x    = (float*)(ws + off);    off += (size_t)NROWS * DMODEL * 4;        // 128 MB
    float*    hg   = (float*)(ws + off);    off += (size_t)NROWS * 2 * DMODEL * 4;    // 256 MB
    _Float16* h16  = (_Float16*)(ws + off); off += (size_t)NROWS * DMODEL * 2;        // 64 MB
    _Float16* wt16 = (_Float16*)(ws + off); off += (size_t)NLAYER * 2 * DMODEL * DMODEL * 2;
    _Float16* e16  = (_Float16*)(ws + off); off += (size_t)VOCAB * DMODEL * 2;
    float*    cA   = (float*)(ws + off);    off += (size_t)BATCH * NCHUNK * DMODEL * 4;
    float*    cH   = (float*)(ws + off);    off += (size_t)BATCH * NCHUNK * DMODEL * 4;
    float*    hini = (float*)(ws + off);    off += (size_t)BATCH * NCHUNK * DMODEL * 4;
    (void)ws_size; (void)in_sizes; (void)n_in; (void)out_size;

    // one-time conversions
    {
        int n = NLAYER * 2 * DMODEL * DMODEL;
        mingru_convert_w<<<(n + 255) / 256, 256, 0, stream>>>(W, wt16);
        mingru_convert_emb<<<(VOCAB * DMODEL + 255) / 256, 256, 0, stream>>>(emb, e16);
        mingru_embed<<<(NROWS * (DMODEL / 4) + 255) / 256, 256, 0, stream>>>(tokens, emb, x);
    }

    dim3 gemm_blk(256);
    for (int l = 0; l < NLAYER; ++l) {
        mingru_ln_f16<<<NROWS, 256, 0, stream>>>(x, ln_g + l * DMODEL, ln_b + l * DMODEL, h16);
        dim3 grid(2 * DMODEL / BN, NROWS / BM);
        mingru_gemm_f16<<<grid, gemm_blk, 0, stream>>>(
            h16, wt16 + (size_t)l * 2 * DMODEL * DMODEL, hg, NROWS, 2 * DMODEL, DMODEL);
        int n1 = BATCH * NCHUNK * DMODEL;
        mingru_scan_p1<<<n1 / 256, 256, 0, stream>>>(hg, cA, cH);
        mingru_scan_p2<<<(BATCH * DMODEL) / 256, 256, 0, stream>>>(cA, cH, hini);
        mingru_scan_p3<<<n1 / 256, 256, 0, stream>>>(hg, hini, x);
    }

    // final norm + logits = xn @ emb^T  (emb f16 row-major is exactly B^T layout)
    mingru_ln_f16<<<NROWS, 256, 0, stream>>>(x, norm_g, norm_b, h16);
    dim3 gridL(VOCAB / BN, NROWS / BM);
    mingru_gemm_f16<<<gridL, gemm_blk, 0, stream>>>(h16, e16, logits, NROWS, VOCAB, DMODEL);
}
